// RationalBitNetAttention_48060684042978
// MI455X (gfx1250) — 16-bit and 8-bit matrix operand order repaired and checked by reading, compiled, not run on hardware
//
#include <hip/hip_runtime.h>
#include <stdint.h>


typedef __attribute__((ext_vector_type(16))) _Float16 v16h;
typedef __attribute__((ext_vector_type(8)))  float    v8f;
typedef __attribute__((ext_vector_type(8)))  int      v8i;
typedef __attribute__((ext_vector_type(4)))  int      v4i;
typedef __attribute__((ext_vector_type(4)))  unsigned v4u;

namespace cfg {
constexpr int B  = 2;
constexpr int S  = 2048;
constexpr int D  = 2048;
constexpr int H  = 16;
constexpr int HD = 128;
constexpr int M  = B * S;
constexpr float QMAX = 127.0f;
constexpr float INV_SQRT_HD = 0.08838834764831845f;
constexpr float NEG_BIG = -3.0e38f;

constexpr size_t OFF_WSUM = 0;
constexpr size_t OFF_WQ   = 256;
constexpr size_t OFF_XQ   = OFF_WQ  + 4ull * D * D;
constexpr size_t OFF_AMX  = OFF_XQ  + (size_t)M * D;
constexpr size_t OFF_QB   = OFF_AMX + (size_t)M * 4;
constexpr size_t OFF_KB   = OFF_QB  + (size_t)M * D * 2;
constexpr size_t OFF_VT   = OFF_KB  + (size_t)M * D * 2;
constexpr size_t OFF_CTX  = OFF_VT  + (size_t)M * D * 2;
constexpr size_t OFF_CQ   = OFF_CTX + (size_t)M * D * 4;
constexpr size_t OFF_AMC  = OFF_CQ  + (size_t)M * D;
constexpr size_t WS_NEED  = OFF_AMC + (size_t)M * 4;
}

using namespace cfg;

union F8  { v8f  v; float    f[8]; };
union I8V { v8i  v; int      i[8]; };
union H16 { v16h v; unsigned u[8]; };

#if defined(__has_builtin)
#if __has_builtin(__builtin_amdgcn_tensor_load_to_lds) && \
    __has_builtin(__builtin_amdgcn_s_wait_tensorcnt)
#define BN_TDM 1
#endif
#endif
#ifndef BN_TDM
#define BN_TDM 0
#endif

__device__ __forceinline__ void bn_tile_issue(int8_t* lds_dst,
                                              const int8_t* gsrc) {
#if BN_TDM
  const unsigned lds_off = (unsigned)(uintptr_t)lds_dst;
  const unsigned long long ga = (unsigned long long)(uintptr_t)gsrc;
  v4u g0;
  g0.x = 1u;
  g0.y = lds_off;
  g0.z = (unsigned)ga;
  g0.w = (unsigned)((ga >> 32) & 0x01FFFFFFu)
         | 0x80000000u;
  v8i g1;
  g1[0] = 0;
  g1[1] = (int)(2048u << 16);
  g1[2] = (int)(2048u << 16);
  g1[3] = (int)(64u << 16);
  g1[4] = 64;
  g1[5] = 2048;
  g1[6] = 0;
  g1[7] = 0;
  const v4i z4 = (v4i){0, 0, 0, 0};
#if __clang_major__ >= 23
  const v8i z8 = (v8i){0, 0, 0, 0, 0, 0, 0, 0};
  __builtin_amdgcn_tensor_load_to_lds(g0, g1, z4, z4, z8, 0);
#else
  __builtin_amdgcn_tensor_load_to_lds(g0, g1, z4, z4, 0);
#endif
#else
  const int lane = threadIdx.x;
#pragma unroll
  for (int c = 0; c < 8; ++c) {
    const int chunk = c * 32 + lane;
    const int row = chunk >> 2, col = (chunk & 3) * 16;
    *(int4*)(lds_dst + row * 64 + col) =
        *(const int4*)(gsrc + (size_t)row * D + col);
  }
#endif
}

template <int N>
__device__ __forceinline__ void bn_tile_wait() {
#if BN_TDM
  __builtin_amdgcn_s_wait_tensorcnt(N);
#endif
  asm volatile("" ::: "memory");
}

__global__ __launch_bounds__(32) void bn_zero_kernel(float* wsum) {
  if (threadIdx.x < 8) wsum[threadIdx.x] = 0.0f;
}

__global__ __launch_bounds__(256) void bn_wabs_kernel(
    const float* __restrict__ w0, const float* __restrict__ w1,
    const float* __restrict__ w2, const float* __restrict__ w3,
    float* __restrict__ wsum) {
  const int wi = blockIdx.y;
  const float* w = (wi == 0) ? w0 : (wi == 1) ? w1 : (wi == 2) ? w2 : w3;
  const size_t base = (size_t)blockIdx.x * 256 * 64;
  float s = 0.0f;
#pragma unroll 8
  for (int i = 0; i < 64; ++i)
    s += __builtin_fabsf(w[base + (size_t)i * 256 + threadIdx.x]);
  __shared__ float red[256];
  red[threadIdx.x] = s;
  __syncthreads();
#pragma unroll
  for (int sft = 128; sft > 0; sft >>= 1) {
    if (threadIdx.x < sft) red[threadIdx.x] += red[threadIdx.x + sft];
    __syncthreads();
  }
  if (threadIdx.x == 0) atomicAdd(&wsum[wi], red[0]);
}

__global__ __launch_bounds__(256) void bn_wquant_kernel(
    const float* __restrict__ w0, const float* __restrict__ w1,
    const float* __restrict__ w2, const float* __restrict__ w3,
    const float* __restrict__ wsum, int8_t* __restrict__ wq8) {
  const int wi = blockIdx.y;
  const float* w = (wi == 0) ? w0 : (wi == 1) ? w1 : (wi == 2) ? w2 : w3;
  const float sc = fmaxf(wsum[wi] * (1.0f / (float)((size_t)D * D)), 1e-6f);
  const size_t idx = (size_t)blockIdx.x * 256 + threadIdx.x;
  float v = rintf(w[idx] / sc);
  v = fminf(fmaxf(v, -1.0f), 1.0f);
  wq8[(size_t)wi * D * D + idx] = (int8_t)v;
}

__global__ __launch_bounds__(256) void bn_aquant_kernel(
    const float* __restrict__ x, int8_t* __restrict__ q,
    float* __restrict__ amax) {
  const int row = blockIdx.x;
  const int tid = threadIdx.x;
  const float* xr = x + (size_t)row * D;
  float m = 0.0f;
  for (int c = tid; c < D; c += 256) m = fmaxf(m, __builtin_fabsf(xr[c]));
  __shared__ float red[256];
  red[tid] = m;
  __syncthreads();
#pragma unroll
  for (int sft = 128; sft > 0; sft >>= 1) {
    if (tid < sft) red[tid] = fmaxf(red[tid], red[tid + sft]);
    __syncthreads();
  }
  const float am = fmaxf(red[0], 1e-8f);
  if (tid == 0) amax[row] = am;
  const float sc = QMAX / am;
  int8_t* qr = q + (size_t)row * D;
  for (int c = tid; c < D; c += 256) {
    float v = rintf(xr[c] * sc);
    v = fminf(fmaxf(v, -QMAX), QMAX);
    qr[c] = (int8_t)v;
  }
}

__global__ __launch_bounds__(32) void bn_gemm_iu8_kernel(
    const int8_t* __restrict__ aq, const int8_t* __restrict__ wq,
    const float* __restrict__ amax, const float* __restrict__ wsum,
    void* __restrict__ outp, int mode) {
  const int lane = threadIdx.x;
  const int half = lane >> 4;
  const int llo  = lane & 15;
  const int n0 = blockIdx.x * 64;
  const int m0 = blockIdx.y * 16;

  __shared__ int8_t btile[2][64 * 64];

  const int8_t* arow  = aq + (size_t)(m0 + llo) * D;
  const int8_t* wbase = wq + (size_t)n0 * D;

  I8V acc[4];
#pragma unroll
  for (int t = 0; t < 4; ++t)
#pragma unroll
    for (int r = 0; r < 8; ++r) acc[t].i[r] = 0;

  bn_tile_issue(btile[0], wbase);

  for (int ks = 0; ks < D / 64; ++ks) {
    const int k = ks * 64;
    const int cur = ks & 1;
    __builtin_prefetch(arow + k + 64, 0, 3);

    I8V a;
#pragma unroll
    for (int v = 0; v < 8; ++v) {
      const int ka =
          k + 32 * (v >> 2) + 16 * ((v >> 1) & 1) + 8 * half + 4 * (v & 1);
      a.i[v] = *(const int*)(arow + ka);
    }

    if (ks + 1 < D / 64) {
      bn_tile_issue(btile[cur ^ 1], wbase + k + 64);
      bn_tile_wait<1>();
    } else {
      bn_tile_wait<0>();
    }

#pragma unroll
    for (int t = 0; t < 4; ++t) {
      I8V b;
#pragma unroll
      for (int v = 0; v < 8; ++v) {
        const int kb = 32 * (v >> 2) + 16 * ((v >> 1) & 1) + 8 * half + 4 * (v & 1);
        b.i[v] = *(const int*)(&btile[cur][(16 * t + llo) * 64 + kb]);
      }
      acc[t].v = __builtin_amdgcn_wmma_i32_16x16x64_iu8(
           true, a.v,  true, b.v, acc[t].v,
           false,  false);
    }
    __syncthreads();
  }

  const float wscale = fmaxf(wsum[0] * (1.0f / (float)((size_t)D * D)), 1e-6f);
#pragma unroll
  for (int t = 0; t < 4; ++t)
#pragma unroll
    for (int r = 0; r < 8; ++r) {
      const int row = m0 + r + 8 * half;
      const int col = n0 + 16 * t + llo;
      const float y = (float)acc[t].i[r] * (wscale * amax[row] * (1.0f / QMAX));
      if (mode == 0) {
        ((_Float16*)outp)[(size_t)row * D + col] = (_Float16)y;
      } else if (mode == 1) {
        const int b_ = row >> 11, s_ = row & (S - 1);
        const int h_ = col >> 7,  d_ = col & (HD - 1);
        ((_Float16*)outp)[((size_t)(b_ * H + h_) * HD + d_) * S + s_] =
            (_Float16)y;
      } else {
        ((float*)outp)[(size_t)row * D + col] = y;
      }
    }
}

__global__ __launch_bounds__(32) void bn_flash_kernel(
    const _Float16* __restrict__ qb, const _Float16* __restrict__ kb,
    const _Float16* __restrict__ vT, float* __restrict__ ctx) {
  const int lane = threadIdx.x;
  const int half = lane >> 4;
  const int llo  = lane & 15;
  const int bh = blockIdx.y;
  const int b  = bh / H;
  const int h  = bh % H;
  const int q0 = blockIdx.x * 16;
  const size_t tokbase = (size_t)b * S;
  const size_t hoff = (size_t)h * HD;

  __shared__ _Float16 pbuf[16 * 32];

  H16 qa[4];
#pragma unroll
  for (int dc = 0; dc < 4; ++dc)
#pragma unroll
    for (int v = 0; v < 8; ++v) {
      const int d = dc * 32 + 16 * (v >> 2) + 8 * half + 2 * (v & 3);
      qa[dc].u[v] =
          *(const unsigned*)(qb + (tokbase + q0 + llo) * D + hoff + d);
    }

  F8 o[8];
#pragma unroll
  for (int t = 0; t < 8; ++t)
#pragma unroll
    for (int r = 0; r < 8; ++r) o[t].f[r] = 0.0f;
  float mrow[8], lrow[8];
#pragma unroll
  for (int r = 0; r < 8; ++r) { mrow[r] = NEG_BIG; lrow[r] = 0.0f; }

  for (int kc = 0; kc < q0 + 16; kc += 32) {
    F8 s0, s1;
#pragma unroll
    for (int r = 0; r < 8; ++r) { s0.f[r] = 0.0f; s1.f[r] = 0.0f; }
#pragma unroll
    for (int dc = 0; dc < 4; ++dc) {
      H16 bk0, bk1;
#pragma unroll
      for (int v = 0; v < 8; ++v) {
        const int d = dc * 32 + 16 * (v >> 2) + 8 * half + 2 * (v & 3);
        bk0.u[v] =
            *(const unsigned*)(kb + (tokbase + kc + llo) * D + hoff + d);
        bk1.u[v] =
            *(const unsigned*)(kb + (tokbase + kc + 16 + llo) * D + hoff + d);
      }
      s0.v = __builtin_amdgcn_wmma_f32_16x16x32_f16(
          false, qa[dc].v, false, bk0.v, (short)0, s0.v, false, false);
      s1.v = __builtin_amdgcn_wmma_f32_16x16x32_f16(
          false, qa[dc].v, false, bk1.v, (short)0, s1.v, false, false);
    }

    const int col0 = kc + llo;
    const int col1 = kc + 16 + llo;
#pragma unroll
    for (int r = 0; r < 8; ++r) {
      const int rowg = q0 + r + 8 * half;
      float a0 = s0.f[r] * INV_SQRT_HD;
      float a1 = s1.f[r] * INV_SQRT_HD;
      if (col0 > rowg) a0 = NEG_BIG;
      if (col1 > rowg) a1 = NEG_BIG;
      float cm = fmaxf(a0, a1);
#pragma unroll
      for (int msk = 1; msk < 16; msk <<= 1)
        cm = fmaxf(cm, __shfl_xor(cm, msk, 32));
      const float mnew = fmaxf(mrow[r], cm);
      const float alpha = __expf(mrow[r] - mnew);
      const float p0 = __expf(a0 - mnew);
      const float p1 = __expf(a1 - mnew);
      float rs = p0 + p1;
#pragma unroll
      for (int msk = 1; msk < 16; msk <<= 1)
        rs += __shfl_xor(rs, msk, 32);
      lrow[r] = lrow[r] * alpha + rs;
      mrow[r] = mnew;
#pragma unroll
      for (int t = 0; t < 8; ++t) o[t].f[r] *= alpha;
      pbuf[(r + 8 * half) * 32 + llo]      = (_Float16)p0;
      pbuf[(r + 8 * half) * 32 + 16 + llo] = (_Float16)p1;
    }
    __syncthreads();

    H16 ap;
#pragma unroll
    for (int v = 0; v < 8; ++v) {
      const int kk = 16 * (v >> 2) + 8 * half + 2 * (v & 3);
      ap.u[v] = *(const unsigned*)&pbuf[llo * 32 + kk];
    }

#pragma unroll
    for (int t = 0; t < 8; ++t) {
      H16 bv;
#pragma unroll
      for (int v = 0; v < 8; ++v) {
        const int key = kc + 16 * (v >> 2) + 8 * half + 2 * (v & 3);
        const int d   = 16 * t + llo;
        bv.u[v] =
            *(const unsigned*)(vT + ((size_t)bh * HD + d) * S + key);
      }
      o[t].v = __builtin_amdgcn_wmma_f32_16x16x32_f16(
          false, ap.v, false, bv.v, (short)0, o[t].v, false, false);
    }
    __syncthreads();
  }

#pragma unroll
  for (int r = 0; r < 8; ++r) {
    const int rowg = q0 + r + 8 * half;
    const float inv = 1.0f / lrow[r];
#pragma unroll
    for (int t = 0; t < 8; ++t)
      ctx[(tokbase + rowg) * D + hoff + 16 * t + llo] = o[t].f[r] * inv;
  }
}

extern "C" void kernel_launch(void* const* d_in, const int* in_sizes, int n_in,
                              void* d_out, int out_size, void* d_ws,
                              size_t ws_size, hipStream_t stream) {
  (void)in_sizes; (void)n_in; (void)out_size; (void)ws_size;
  const float* x  = (const float*)d_in[0];
  const float* wq = (const float*)d_in[1];
  const float* wk = (const float*)d_in[2];
  const float* wv = (const float*)d_in[3];
  const float* wo = (const float*)d_in[4];

  char* ws = (char*)d_ws;
  float*    wsum = (float*)   (ws + OFF_WSUM);
  int8_t*   wq8  = (int8_t*)  (ws + OFF_WQ);
  int8_t*   xq   = (int8_t*)  (ws + OFF_XQ);
  float*    amx  = (float*)   (ws + OFF_AMX);
  _Float16* qb   = (_Float16*)(ws + OFF_QB);
  _Float16* kb   = (_Float16*)(ws + OFF_KB);
  _Float16* vT   = (_Float16*)(ws + OFF_VT);
  float*    ctx  = (float*)   (ws + OFF_CTX);
  int8_t*   cq   = (int8_t*)  (ws + OFF_CQ);
  float*    amc  = (float*)   (ws + OFF_AMC);

  bn_zero_kernel<<<1, 32, 0, stream>>>(wsum);
  bn_wabs_kernel<<<dim3(256, 4), 256, 0, stream>>>(wq, wk, wv, wo, wsum);
  bn_wquant_kernel<<<dim3(D * D / 256, 4), 256, 0, stream>>>(wq, wk, wv, wo,
                                                             wsum, wq8);
  bn_aquant_kernel<<<M, 256, 0, stream>>>(x, xq, amx);

  const dim3 ggrid(D / 64, M / 16);
  bn_gemm_iu8_kernel<<<ggrid, 32, 0, stream>>>(xq, wq8 + 0 * (size_t)D * D,
                                               amx, wsum + 0, qb, 0);
  bn_gemm_iu8_kernel<<<ggrid, 32, 0, stream>>>(xq, wq8 + 1 * (size_t)D * D,
                                               amx, wsum + 1, kb, 0);
  bn_gemm_iu8_kernel<<<ggrid, 32, 0, stream>>>(xq, wq8 + 2 * (size_t)D * D,
                                               amx, wsum + 2, vT, 1);
  bn_flash_kernel<<<dim3(S / 16, B * H), 32, 0, stream>>>(qb, kb, vT, ctx);
  bn_aquant_kernel<<<M, 256, 0, stream>>>(ctx, cq, amc);
  bn_gemm_iu8_kernel<<<ggrid, 32, 0, stream>>>(cq, wq8 + 3 * (size_t)D * D,
                                               amc, wsum + 3, d_out, 2);
}
